// UltraLSNTForecaster_87875030876722
// MI455X (gfx1250) — matrix operand order repaired and checked by reading, not compiled, not run on hardware
//
#include <hip/hip_runtime.h>
#include <hip/hip_bf16.h>

typedef _Float16 f16;
typedef __attribute__((ext_vector_type(16))) _Float16 v16h;
typedef __attribute__((ext_vector_type(8)))  _Float16 h8v;
typedef __attribute__((ext_vector_type(8)))  float    v8f;

#define BB 2048
#define SS 96
#define DD 256
#define EE 8
#define PDEC 24
#define BSR (BB*SS)

__device__ __forceinline__ float geluf(float x){ return 0.5f*x*(1.0f+erff(x*0.70710678118654752f)); }
__device__ __forceinline__ float sigm(float x){ return 1.0f/(1.0f+expf(-x)); }
__device__ __forceinline__ float softp(float x){ return fmaxf(x,0.0f)+log1pf(expf(-fabsf(x))); }
__device__ __forceinline__ v8f vzero8(){ v8f z; for(int i=0;i<8;i++) z[i]=0.0f; return z; }
__device__ __forceinline__ h8v hzero8(){ h8v z; for(int i=0;i<8;i++) z[i]=(f16)0.0f; return z; }

__device__ __forceinline__ v8f wmmaf16(v16h a, v16h b, v8f c){
  return __builtin_amdgcn_wmma_f32_16x16x32_f16(false, a, false, b, (short)0, c, false, false);
}

// A fragment (16x32 f16) from natural row-major LDS tile:
// lane's 16 values are K = {lh*8..+7, 16+lh*8..+7} -> two ds_load_b128.
__device__ __forceinline__ v16h frag_a(const f16* lds, int stride, int row, int kbase, int lh){
  const f16* p = lds + row*stride + kbase + lh*8;
  h8v lo = *(const h8v*)(p);
  h8v hi = *(const h8v*)(p + 16);
  return __builtin_shufflevector(lo, hi, 0,1,2,3,4,5,6,7,8,9,10,11,12,13,14,15);
}
// B fragment (32x16 f16) from COLUMN-MAJOR staged LDS tile Bt[col][k]:
// lane's 16 values are K = {lh*8..+7, 16+lh*8..+7} at fixed col -> two ds_load_b128.
__device__ __forceinline__ v16h frag_bt(const f16* lds, int stride, int col, int kbase, int lh){
  const f16* p = lds + col*stride + kbase + lh*8;
  h8v lo = *(const h8v*)(p);
  h8v hi = *(const h8v*)(p + 16);
  return __builtin_shufflevector(lo, hi, 0,1,2,3,4,5,6,7,8,9,10,11,12,13,14,15);
}

// CDNA5 async gather into LDS (GLOBAL_LOAD_ASYNC_TO_LDS_B128, ASYNCcnt-tracked)
__device__ __forceinline__ void async_lds_b128(const void* lds_ptr, const void* gptr){
  unsigned l = (unsigned)(unsigned long long)(uintptr_t)lds_ptr;   // LDS aperture: low 32 bits = LDS offset
  unsigned long long a = (unsigned long long)(uintptr_t)gptr;
  asm volatile("global_load_async_to_lds_b128 %0, %1, off" :: "v"(l), "v"(a) : "memory");
}
__device__ __forceinline__ void wait_async(){
  asm volatile("s_wait_asynccnt 0" ::: "memory");
}

// ---------------- generic fused GEMM, double-buffered / async-pipelined ----------------
struct GemmArgs {
  const f16* A;  long long lda, bA;
  const f16* Bw; long long ldb, bB;
  const float* bias; long long bBias;
  const float* resid; long long ldr, bR; int rmod;   // rmod!=0 -> resid row = m % rmod (pos broadcast)
  float* oF; long long ldcF, bCF;
  f16*   oH; long long ldcH, bCH;
  int M, N, K, act;        // act: 0 none, 1 gelu(erf)
  int imS, imC;            // imS!=0 -> A is [B,imS,256] f16, im2col window 3 along S (K=768, imC==256)
};

#define LDA_S 40   // f16 stride (80B, 16B-aligned rows)
#define LDB_T 40   // f16 stride for transposed B tile

__global__ void __launch_bounds__(256) gemm_f16_kernel(GemmArgs g){
  __shared__ __align__(16) f16 As[2][128*LDA_S];
  __shared__ __align__(16) f16 Bt[2][64*LDB_T];
  const int tid = threadIdx.x;
  const int z = blockIdx.z;
  const f16* A  = g.A  + (size_t)z*g.bA;
  const f16* Bw = g.Bw + (size_t)z*g.bB;
  const int m0 = blockIdx.x*128, n0 = blockIdx.y*64;
  const int wave = tid>>5, lane = tid&31, r = lane&15, lh = lane>>4;
  const int mw = wave&3, nw = wave>>2;           // 4 waves along M, 2 along N
  const int arow = tid>>2, acol = (tid&3)*8;
  const int brow = tid>>3, bcol = (tid&7)*8;
  const bool plain = (g.imS==0);
  v8f c00=vzero8(), c01=vzero8(), c10=vzero8(), c11=vzero8();

  // hoist im2col row decomposition (loop-invariant)
  int bi0=0, si0=0, bi1=0, si1=0;
  if(!plain){
    int m = m0 + arow;      bi0 = m/g.imS; si0 = m - bi0*g.imS;
    m += 64;                bi1 = m/g.imS; si1 = m - bi1*g.imS;
  }

  auto issueA_async = [&](int k0, int buf){
    async_lds_b128(&As[buf][arow*LDA_S + acol],      A + (size_t)(m0+arow)*g.lda + k0 + acol);
    async_lds_b128(&As[buf][(arow+64)*LDA_S + acol], A + (size_t)(m0+arow+64)*g.lda + k0 + acol);
  };
  auto loadA_im2col = [&](int k0, h8v& v0, h8v& v1){
    int t  = (k0 + acol) >> 8;           // imC == 256, 8-wide block never crosses a tap
    int ch = (k0 + acol) & 255;
    int sx0 = si0 + t - 1, sx1 = si1 + t - 1;
    v0 = hzero8(); v1 = hzero8();
    if(sx0>=0 && sx0<g.imS) v0 = *(const h8v*)(A + ((size_t)bi0*g.imS + sx0)*256 + ch);
    if(sx1>=0 && sx1<g.imS) v1 = *(const h8v*)(A + ((size_t)bi1*g.imS + sx1)*256 + ch);
  };
  auto storeA = [&](h8v v0, h8v v1, int buf){
    *(h8v*)&As[buf][arow*LDA_S + acol]      = v0;
    *(h8v*)&As[buf][(arow+64)*LDA_S + acol] = v1;
  };
  auto loadB = [&](int k0)->h8v{
    return *(const h8v*)(Bw + (size_t)(k0+brow)*g.ldb + n0 + bcol);
  };
  auto storeB = [&](h8v v, int buf){
#pragma unroll
    for(int i=0;i<8;i++) Bt[buf][(bcol+i)*LDB_T + brow] = v[i];
  };

  // prologue: stage tile k0=0 into buffer 0
  if(plain) issueA_async(0, 0);
  else { h8v a0,a1; loadA_im2col(0, a0, a1); storeA(a0, a1, 0); }
  storeB(loadB(0), 0);
  if(plain) wait_async();
  __syncthreads();

  int cur = 0;
  for(int k0=0; k0<g.K; k0+=32){
    const int nxt = cur^1;
    const bool more = (k0+32) < g.K;
    h8v vb = hzero8(), va0 = hzero8(), va1 = hzero8();
    if(more){
      // stage tile i+1: async engine fills As[nxt] while we compute on [cur]
      if(plain) issueA_async(k0+32, nxt);
      else      loadA_im2col(k0+32, va0, va1);
      vb = loadB(k0+32);
      if(k0+64 < g.K)
        __builtin_prefetch((const void*)(Bw + (size_t)(k0+64+brow)*g.ldb + n0 + bcol), 0, 0);
    }

    v16h a0 = frag_a (&As[cur][0], LDA_S, mw*32 + r,      0, lh);
    v16h a1 = frag_a (&As[cur][0], LDA_S, mw*32 + 16 + r, 0, lh);
    v16h b0 = frag_bt(&Bt[cur][0], LDB_T, nw*32 + r,      0, lh);
    v16h b1 = frag_bt(&Bt[cur][0], LDB_T, nw*32 + 16 + r, 0, lh);
    c00 = wmmaf16(a0,b0,c00);
    c01 = wmmaf16(a0,b1,c01);
    c10 = wmmaf16(a1,b0,c10);
    c11 = wmmaf16(a1,b1,c11);

    if(more){
      if(!plain) storeA(va0, va1, nxt);
      storeB(vb, nxt);
      if(plain) wait_async();
    }
    __syncthreads();
    cur = nxt;
  }

  const float* bias  = g.bias  ? g.bias  + (size_t)z*g.bBias : nullptr;
  const float* resid = g.resid ? g.resid + (size_t)z*g.bR    : nullptr;
  float* oF = g.oF ? g.oF + (size_t)z*g.bCF : nullptr;
  f16*   oH = g.oH ? g.oH + (size_t)z*g.bCH : nullptr;

  auto store = [&](v8f acc, int mb, int nb){
    int n = nb + r;
    float bv = bias ? bias[n] : 0.0f;
#pragma unroll
    for(int v=0; v<8; v++){
      int m = mb + lh*8 + v;
      float val = acc[v] + bv;
      if(resid){
        long long rrow = g.rmod ? (long long)(m % g.rmod) : (long long)m;
        val += resid[rrow*g.ldr + n];
      }
      if(g.act==1) val = geluf(val);
      if(oF) oF[(size_t)m*g.ldcF + n] = val;
      if(oH) oH[(size_t)m*g.ldcH + n] = (f16)val;
    }
  };
  store(c00, m0+mw*32,    n0+nw*32);
  store(c01, m0+mw*32,    n0+nw*32+16);
  store(c10, m0+mw*32+16, n0+nw*32);
  store(c11, m0+mw*32+16, n0+nw*32+16);
}

// ---------------- attention: one (batch,head) per workgroup, fully LDS resident ---------
#define AQ_S 72    // Q natural [s][dh]
#define AK_S 72    // K natural [s][dh]  (this IS col-major for the Q@K^T B-operand)
#define AV_T 104   // V transposed [dh][s]
#define AP_S 104   // P natural [s][s]

__global__ void __launch_bounds__(192) attn_kernel(const f16* __restrict__ Qg, const f16* __restrict__ Kg,
                                                   const f16* __restrict__ Vg, f16* __restrict__ Og){
  __shared__ __align__(16) f16 Qs[96*AQ_S];
  __shared__ __align__(16) f16 Ks[96*AK_S];
  __shared__ __align__(16) f16 Vt[64*AV_T];
  __shared__ __align__(16) f16 Pb[96*AP_S];
  const int tid = threadIdx.x;
  const int b = blockIdx.x >> 2, hd = blockIdx.x & 3;
  const size_t base = (size_t)b*96*256 + (size_t)hd*64;
  const int row = tid>>1, cbo = (tid&1)*32;
#pragma unroll
  for(int c=0;c<32;c+=8){
    async_lds_b128(&Qs[row*AQ_S+cbo+c], Qg + base + (size_t)row*256 + cbo + c);
    async_lds_b128(&Ks[row*AK_S+cbo+c], Kg + base + (size_t)row*256 + cbo + c);
  }
  {
    // V: load rows, scatter transposed -> Vt[dh][s]
#pragma unroll
    for(int c=0;c<32;c+=8){
      h8v v = *(const h8v*)(Vg + base + (size_t)row*256 + cbo + c);
#pragma unroll
      for(int i=0;i<8;i++) Vt[(cbo+c+i)*AV_T + row] = v[i];
    }
  }
  wait_async();
  __syncthreads();

  const int wave=tid>>5, lane=tid&31, r=lane&15, lh=lane>>4;
  v8f sacc[6];
  for(int nt=0; nt<6; nt++){
    v8f acc = vzero8();
#pragma unroll
    for(int ks=0; ks<64; ks+=32){
      v16h a  = frag_a (Qs, AQ_S, wave*16 + r, ks, lh);
      v16h bb = frag_bt(Ks, AK_S, nt*16 + r, ks, lh);
      acc = wmmaf16(a, bb, acc);
    }
    sacc[nt] = acc;
  }
  // softmax over the 96-wide rows held across 16 lanes (one half) x 6 fragments
#pragma unroll
  for(int v=0; v<8; v++){
    float mx = -3.0e38f;
#pragma unroll
    for(int nt=0; nt<6; nt++) mx = fmaxf(mx, sacc[nt][v]*0.125f);
    for(int msk=1; msk<16; msk<<=1) mx = fmaxf(mx, __shfl_xor(mx, msk, 32));
    float sm = 0.0f;
#pragma unroll
    for(int nt=0; nt<6; nt++){ float e = expf(sacc[nt][v]*0.125f - mx); sacc[nt][v]=e; sm+=e; }
    for(int msk=1; msk<16; msk<<=1) sm += __shfl_xor(sm, msk, 32);
    float inv = 1.0f/sm;
    int rw = wave*16 + lh*8 + v;
#pragma unroll
    for(int nt=0; nt<6; nt++) Pb[rw*AP_S + nt*16 + r] = (f16)(sacc[nt][v]*inv);
  }
  __syncthreads();
  // O = P @ V
  for(int nt=0; nt<4; nt++){
    v8f acc = vzero8();
#pragma unroll
    for(int ks=0; ks<96; ks+=32){
      v16h a  = frag_a (Pb + wave*16*AP_S, AP_S, r, ks, lh);
      v16h bb = frag_bt(Vt, AV_T, nt*16 + r, ks, lh);
      acc = wmmaf16(a, bb, acc);
    }
#pragma unroll
    for(int v=0; v<8; v++)
      Og[base + (size_t)(wave*16 + lh*8 + v)*256 + nt*16 + r] = (f16)acc[v];
  }
}

// ---------------- LayerNorm over D=256, one wave per row ------------------------------
__global__ void __launch_bounds__(256) ln_kernel(const float* __restrict__ x, const float* __restrict__ gg,
                                                 const float* __restrict__ bb, float* oF, f16* oH, int rows){
  int wave = threadIdx.x>>5, lane = threadIdx.x&31;
  int rowi = blockIdx.x*8 + wave;
  if(rowi>=rows) return;
  const float* xr = x + (size_t)rowi*256;
  float vals[8]; float s=0.0f;
#pragma unroll
  for(int i=0;i<8;i++){ vals[i]=xr[lane+32*i]; s+=vals[i]; }
  for(int msk=16; msk>=1; msk>>=1) s += __shfl_xor(s, msk, 32);
  float mean = s*(1.0f/256.0f);
  float q=0.0f;
#pragma unroll
  for(int i=0;i<8;i++){ float d=vals[i]-mean; q+=d*d; }
  for(int msk=16; msk>=1; msk>>=1) q += __shfl_xor(q, msk, 32);
  float inv = rsqrtf(q*(1.0f/256.0f) + 1e-5f);
#pragma unroll
  for(int i=0;i<8;i++){
    int c = lane+32*i;
    float o = (vals[i]-mean)*inv*gg[c] + bb[c];
    if(oF) oF[(size_t)rowi*256 + c] = o;
    if(oH) oH[(size_t)rowi*256 + c] = (f16)o;
  }
}

// ---------------- utility kernels ------------------------------------------------------
__global__ void cvt_f32_f16(const float* __restrict__ s, f16* __restrict__ d, long long n){
  long long i = (long long)blockIdx.x*blockDim.x + threadIdx.x;
  if(i<n) d[i] = (f16)s[i];
}
// conv weight (Cout,Cin,3) -> B-matrix [(t*256+ci)][co] f16
__global__ void repack_conv(const float* __restrict__ s, f16* __restrict__ d){
  int i = blockIdx.x*256 + threadIdx.x;
  if(i>=768*256) return;
  int co = i & 255, rest = i >> 8;
  int ci = rest & 255, t = rest >> 8;
  d[i] = (f16)s[(co*256 + ci)*3 + t];
}
__global__ void mean_reduce(const float* __restrict__ x, float* out, int n){
  __shared__ float red[256];
  float a=0.0f;
  for(int i=threadIdx.x;i<n;i+=256) a+=x[i];
  red[threadIdx.x]=a; __syncthreads();
  for(int st=128; st>0; st>>=1){ if(threadIdx.x<st) red[threadIdx.x]+=red[threadIdx.x+st]; __syncthreads(); }
  if(threadIdx.x==0) out[0] = red[0]/(float)n;
}
// tiny 2-layer MLP head per row: out = act2( gelu(in@W1+b1) @ W2 + b2 )
__global__ void mlp2_kernel(const float* __restrict__ in1, int d1, const float* __restrict__ in2, int d2,
                            const float* __restrict__ W1, const float* __restrict__ b1, int dh,
                            const float* __restrict__ W2, const float* __restrict__ b2,
                            float* out, int n, int act2){ // act2: 0 sigmoid, 1 softplus
  int b = blockIdx.x*blockDim.x + threadIdx.x;
  if(b>=n) return;
  float hid[64];
  for(int j=0;j<dh;j++) hid[j]=b1[j];
  const float* x1 = in1 + (size_t)b*d1;
  for(int d=0; d<d1; d++){ float v=x1[d]; const float* w=W1+(size_t)d*dh;
    for(int j=0;j<dh;j++) hid[j]+=v*w[j]; }
  if(in2){
    const float* x2 = in2 + (size_t)b*d2;
    for(int d=0; d<d2; d++){ float v=x2[d]; const float* w=W1+(size_t)(d1+d)*dh;
      for(int j=0;j<dh;j++) hid[j]+=v*w[j]; }
  }
  float o=b2[0];
  for(int j=0;j<dh;j++) o += geluf(hid[j])*W2[j];
  out[b] = act2 ? softp(o) : sigm(o);
}
__global__ void router_kernel(const float* __restrict__ h, const float* __restrict__ unc,
                              const float* __restrict__ um, const float* __restrict__ rw,
                              float* __restrict__ cw, int n){
  int b = blockIdx.x*blockDim.x + threadIdx.x;
  if(b>=n) return;
  float un = unc[b]/(um[0]+1e-8f);
  float lg[8];
#pragma unroll
  for(int e=0;e<8;e++) lg[e]= un*rw[256*8+e];
  const float* hr = h + (size_t)b*256;
  for(int d=0; d<256; d++){
    float hv=hr[d]; const float* rr = rw + d*8;
#pragma unroll
    for(int e=0;e<8;e++) lg[e]+= hv*rr[e];
  }
  float mx=lg[0];
  for(int e=1;e<8;e++) mx=fmaxf(mx,lg[e]);
  float pp[8];
  for(int e=0;e<8;e++) pp[e]=expf(lg[e]-mx);
  bool used[8]={false,false,false,false,false,false,false,false};
  float s=0.0f;
  for(int kk=0;kk<4;kk++){
    int best=0; float bv=-1.0f;
    for(int e=0;e<8;e++) if(!used[e] && pp[e]>bv){ bv=pp[e]; best=e; }
    used[best]=true; s+=bv;
  }
  float invs=1.0f/s;
  for(int e=0;e<8;e++) cw[b*8+e] = used[e] ? pp[e]*invs : 0.0f;
}
__global__ void moe_combine_kernel(float* h, f16* hh, const float* __restrict__ eo,
                                   const float* __restrict__ cw, int n){
  int i = blockIdx.x*blockDim.x + threadIdx.x;
  if(i>=n*256) return;
  int b=i>>8, d=i&255;
  float a=h[i];
#pragma unroll
  for(int e=0;e<8;e++) a += cw[b*8+e]*eo[((size_t)b*8+e)*256 + d];
  h[i]=a; hh[i]=(f16)a;
}
__global__ void mode_select_kernel(const float* __restrict__ mode, const float* __restrict__ fastv,
                                   const float* __restrict__ slowv, float* outb, int n){
  int i = blockIdx.x*blockDim.x + threadIdx.x;
  if(i>=n*256) return;
  int b=i>>8;
  outb[i] = (mode[b]>0.5f) ? slowv[i] : fastv[i];
}
__global__ void skip_kernel(float* h, f16* hh, const float* __restrict__ outb,
                            const float* __restrict__ imp, const float* __restrict__ chg,
                            const float* __restrict__ thr, int n){
  int i = blockIdx.x*blockDim.x + threadIdx.x;
  if(i>=n*256) return;
  int b=i>>8;
  float sc=(1.0f-imp[b])*(1.0f-chg[b]);
  float v = (sc > sigm(thr[0])) ? h[i] : outb[i];
  h[i]=v; hh[i]=(f16)v;
}
__global__ void head2_kernel(const f16* __restrict__ f1, const float* __restrict__ w,
                             const float* __restrict__ b2, float* out, int n){
  int i = blockIdx.x*blockDim.x + threadIdx.x;
  if(i>=n) return;
  const f16* row = f1 + (size_t)i*128;
  float a=b2[0];
  for(int j=0;j<128;j++) a += (float)row[j]*w[j];
  out[i]=a;
}

// -------------------------------------------------------------------------------------
extern "C" void kernel_launch(void* const* d_in, const int* in_sizes, int n_in,
                              void* d_out, int out_size, void* d_ws, size_t ws_size,
                              hipStream_t stream){
  (void)in_sizes; (void)n_in; (void)out_size; (void)ws_size;
  auto F32 = [&](int i)->const float*{ return (const float*)d_in[i]; };

  // ---- bump allocator over d_ws (~1.2 GB total) ----
  char* wsb = (char*)d_ws; size_t off = 0;
  auto alloc = [&](size_t bytes)->void*{ void* p = wsb + off; off += (bytes + 255) & ~(size_t)255; return p; };

  // f16 weights
  f16* xh    = (f16*)alloc((size_t)BSR*64*2);
  f16* wproj = (f16*)alloc(64*256*2);
  f16* wc1   = (f16*)alloc(768*256*2);
  f16* wc2   = (f16*)alloc(768*256*2);
  f16* wqh   = (f16*)alloc(65536*2);
  f16* wkh   = (f16*)alloc(65536*2);
  f16* wvh   = (f16*)alloc(65536*2);
  f16* woh   = (f16*)alloc(65536*2);
  f16* wagg1 = (f16*)alloc((size_t)24576*512*2);
  f16* wagg2 = (f16*)alloc(131072*2);
  f16 *wup[4], *wdn[4], *wfa[4], *ws1[4], *ws2[4];
  for(int i=0;i<4;i++){
    wup[i]=(f16*)alloc((size_t)8*256*1024*2);
    wdn[i]=(f16*)alloc((size_t)8*1024*256*2);
    wfa[i]=(f16*)alloc(65536*2);
    ws1[i]=(f16*)alloc(262144*2);
    ws2[i]=(f16*)alloc(262144*2);
  }
  f16* wexp = (f16*)alloc((size_t)256*6144*2);
  f16* whd1 = (f16*)alloc(32768*2);
  // activations
  float* G1 = (float*)alloc((size_t)BSR*256*4);
  float* G2 = (float*)alloc((size_t)BSR*256*4);
  f16* F1 = (f16*)alloc((size_t)BSR*256*2);
  f16* F2 = (f16*)alloc((size_t)BSR*256*2);
  f16* F3 = (f16*)alloc((size_t)BSR*256*2);
  f16* F4 = (f16*)alloc((size_t)BSR*256*2);
  f16* F5 = (f16*)alloc((size_t)BSR*256*2);
  f16* F6 = (f16*)alloc((size_t)BSR*256*2);
  float* hB  = (float*)alloc((size_t)BB*256*4);
  f16*   hBh = (f16*)alloc((size_t)BB*256*2);
  f16*   a1h = (f16*)alloc((size_t)BB*512*2);
  float* uncB=(float*)alloc(BB*4);
  float* uncM=(float*)alloc(256);
  float* cwB =(float*)alloc((size_t)BB*8*4);
  f16*   moeH=(f16*)alloc((size_t)BB*8*1024*2);
  float* eoF =(float*)alloc((size_t)BB*8*256*4);
  float* modeB=(float*)alloc(BB*4);
  float* impB =(float*)alloc(BB*4);
  float* chgB =(float*)alloc(BB*4);
  float* tmpF =(float*)alloc((size_t)BB*256*4);
  float* fastF=(float*)alloc((size_t)BB*256*4);
  float* slowF=(float*)alloc((size_t)BB*256*4);
  float* outbF=(float*)alloc((size_t)BB*256*4);
  f16*   smidH=(f16*)alloc((size_t)BB*1024*2);
  f16*   zH   =(f16*)alloc((size_t)BB*6144*2);
  f16*   f1H  =(f16*)alloc((size_t)BB*PDEC*128*2);

  auto cvt = [&](int idx, f16* dst, long long n){
    cvt_f32_f16<<<(unsigned)((n+255)/256), 256, 0, stream>>>(F32(idx), dst, n);
  };
  auto gemm = [&](const f16* A, long long lda, long long bA,
                  const f16* Bw, long long ldb, long long bBm,
                  const float* bias, long long bBias,
                  const float* resid, long long ldr, long long bR, int rmod,
                  float* oF, long long ldcF, long long bCF,
                  f16* oH, long long ldcH, long long bCH,
                  int M, int N, int K, int act, int imS, int imC, int batch){
    GemmArgs g; g.A=A; g.lda=lda; g.bA=bA; g.Bw=Bw; g.ldb=ldb; g.bB=bBm;
    g.bias=bias; g.bBias=bBias; g.resid=resid; g.ldr=ldr; g.bR=bR; g.rmod=rmod;
    g.oF=oF; g.ldcF=ldcF; g.bCF=bCF; g.oH=oH; g.ldcH=ldcH; g.bCH=bCH;
    g.M=M; g.N=N; g.K=K; g.act=act; g.imS=imS; g.imC=imC;
    dim3 gr((unsigned)(M/128), (unsigned)(N/64), (unsigned)batch);
    gemm_f16_kernel<<<gr, 256, 0, stream>>>(g);
  };

  // ---- weight/input conversion (f32 -> f16, per call; order = setup_inputs dict order) ----
  cvt(0, xh, (long long)BSR*64);
  cvt(1, wproj, 64*256);
  repack_conv<<<768,256,0,stream>>>(F32(4), wc1);
  repack_conv<<<768,256,0,stream>>>(F32(6), wc2);
  cvt(8,  wqh, 65536); cvt(10, wkh, 65536); cvt(12, wvh, 65536); cvt(14, woh, 65536);
  cvt(18, wagg1, (long long)24576*512);
  cvt(20, wagg2, 131072);
  for(int bi=0; bi<4; bi++){
    int ib = 22 + 32*bi;
    cvt(ib+5,  wup[bi], (long long)8*256*1024);
    cvt(ib+7,  wdn[bi], (long long)8*1024*256);
    cvt(ib+13, wfa[bi], 65536);
    cvt(ib+17, ws1[bi], 262144);
    cvt(ib+19, ws2[bi], 262144);
  }
  cvt(150, wexp, (long long)256*6144);
  cvt(152, whd1, 32768);

  // ---- encoder ----
  // h0 = x@proj + b + pos  -> G1 (f32) + F1 (f16)
  gemm(xh,64,0, wproj,256,0, F32(2),0, F32(3),256,0,SS, G1,256,0, F1,256,0, BSR,256,64, 0, 0,0, 1);
  // c1 = gelu(conv1(h0)+b)  (implicit im2col K=768) -> F2
  gemm(F1,0,0, wc1,256,0, F32(5),0, nullptr,0,0,0, nullptr,0,0, F2,256,0, BSR,256,768, 1, SS,256, 1);
  // h1 = h0 + conv2(c1)+b -> G2 (f32) + F3 (f16)
  gemm(F2,0,0, wc2,256,0, F32(7),0, G1,256,0,0, G2,256,0, F3,256,0, BSR,256,768, 0, SS,256, 1);
  // q,k,v
  gemm(F3,256,0, wqh,256,0, F32(9),0,  nullptr,0,0,0, nullptr,0,0, F4,256,0, BSR,256,256, 0,0,0,1);
  gemm(F3,256,0, wkh,256,0, F32(11),0, nullptr,0,0,0, nullptr,0,0, F5,256,0, BSR,256,256, 0,0,0,1);
  gemm(F3,256,0, wvh,256,0, F32(13),0, nullptr,0,0,0, nullptr,0,0, F6,256,0, BSR,256,256, 0,0,0,1);
  attn_kernel<<<BB*4, 192, 0, stream>>>(F4, F5, F6, F1);
  // wo(o) + b + h1 -> G1; LN -> F2 (f16 [B, S*D])
  gemm(F1,256,0, woh,256,0, F32(15),0, G2,256,0,0, G1,256,0, nullptr,0,0, BSR,256,256, 0,0,0,1);
  ln_kernel<<<BSR/8, 256, 0, stream>>>(G1, F32(16), F32(17), nullptr, F2, BSR);
  // agg1: [2048,24576]x[24576,512] gelu -> a1h ; agg2 -> hB/hBh
  gemm(F2,24576,0, wagg1,512,0, F32(19),0, nullptr,0,0,0, nullptr,0,0, a1h,512,0, BB,512,24576, 1,0,0,1);
  gemm(a1h,512,0,  wagg2,256,0, F32(21),0, nullptr,0,0,0, hB,256,0, hBh,256,0, BB,256,512, 0,0,0,1);

  // ---- 4 blocks ----
  for(int bi=0; bi<4; bi++){
    int ib = 22 + 32*bi;
    mlp2_kernel<<<BB/128,128,0,stream>>>(hB,256, nullptr,0, F32(ib+0),F32(ib+1),16, F32(ib+2),F32(ib+3), uncB, BB, 1);
    mean_reduce<<<1,256,0,stream>>>(uncB, uncM, BB);
    router_kernel<<<BB/128,128,0,stream>>>(hB, uncB, uncM, F32(ib+4), cwB, BB);
    // dense MoE: all 8 experts, strided-batched over grid.z
    gemm(hBh,256,0,    wup[bi],1024,262144, F32(ib+6),1024, nullptr,0,0,0, nullptr,0,0, moeH,8192,1024, BB,1024,256, 1,0,0, 8);
    gemm(moeH,8192,1024, wdn[bi],256,262144, F32(ib+8),256, nullptr,0,0,0, eoF,2048,256, nullptr,0,0, BB,256,1024, 0,0,0, 8);
    moe_combine_kernel<<<BB,256,0,stream>>>(hB, hBh, eoF, cwB, BB);
    mlp2_kernel<<<BB/128,128,0,stream>>>(hB,256, nullptr,0, F32(ib+9),F32(ib+10),64, F32(ib+11),F32(ib+12), modeB, BB, 0);
    // fast path
    gemm(hBh,256,0, wfa[bi],256,0, F32(ib+14),0, nullptr,0,0,0, tmpF,256,0, nullptr,0,0, BB,256,256, 0,0,0,1);
    ln_kernel<<<BB/8,256,0,stream>>>(tmpF, F32(ib+15), F32(ib+16), fastF, nullptr, BB);
    // slow path
    gemm(hBh,256,0,   ws1[bi],1024,0, F32(ib+18),0, nullptr,0,0,0, nullptr,0,0, smidH,1024,0, BB,1024,256, 1,0,0,1);
    gemm(smidH,1024,0, ws2[bi],256,0, F32(ib+20),0, nullptr,0,0,0, tmpF,256,0, nullptr,0,0, BB,256,1024, 0,0,0,1);
    ln_kernel<<<BB/8,256,0,stream>>>(tmpF, F32(ib+21), F32(ib+22), slowF, nullptr, BB);
    mode_select_kernel<<<BB,256,0,stream>>>(modeB, fastF, slowF, outbF, BB);
    mlp2_kernel<<<BB/128,128,0,stream>>>(hB,256, nullptr,0, F32(ib+23),F32(ib+24),64, F32(ib+25),F32(ib+26), impB, BB, 0);
    mlp2_kernel<<<BB/128,128,0,stream>>>(hB,256, outbF,256, F32(ib+27),F32(ib+28),64, F32(ib+29),F32(ib+30), chgB, BB, 0);
    skip_kernel<<<BB,256,0,stream>>>(hB, hBh, outbF, impB, chgB, F32(ib+31), BB);
  }

  // ---- decoder ----
  gemm(hBh,256,0, wexp,6144,0, F32(151),0, nullptr,0,0,0, nullptr,0,0, zH,6144,0, BB,6144,256, 1,0,0,1);
  gemm(zH,256,0,  whd1,128,0,  F32(153),0, nullptr,0,0,0, nullptr,0,0, f1H,128,0, BB*PDEC,128,256, 1,0,0,1);
  head2_kernel<<<(BB*PDEC)/256,256,0,stream>>>(f1H, F32(154), F32(155), (float*)d_out, BB*PDEC);
}
